// GNN_30331059044726
// MI455X (gfx1250) — matrix operand order repaired and checked by reading, not compiled, not run on hardware
//
#include <hip/hip_runtime.h>
#include <hip/hip_bf16.h>

typedef __attribute__((ext_vector_type(16))) _Float16 v16h;
typedef __attribute__((ext_vector_type(8)))  _Float16 v8h;
typedef __attribute__((ext_vector_type(4)))  _Float16 v4h;
typedef __attribute__((ext_vector_type(8)))  float    v8f;
typedef __attribute__((ext_vector_type(4)))  float    v4f;

#define EPS_BN 1e-5f

// ---------------------------------------------------------------------------
// Utility kernels
// ---------------------------------------------------------------------------
__global__ void zero_f32(float* __restrict__ p, size_t n) {
    size_t i = (size_t)blockIdx.x * blockDim.x + threadIdx.x;
    if (i < n) p[i] = 0.0f;
}

// T2 = -0.5 * H   (Cheb recurrence init; consuming GEMM applies ascale=2)
__global__ void scale_init(const float* __restrict__ h, float* __restrict__ t2, size_t n) {
    size_t i = (size_t)blockIdx.x * blockDim.x + threadIdx.x;
    if (i < n) t2[i] = -0.5f * h[i];
}

// Xp[i*32 + c] = (c < 16) ? x[i*16 + c] : 0    (pad K to 32 for encoder-0)
__global__ void pad_x(const float* __restrict__ x, float* __restrict__ xp, int Mn) {
    int i = blockIdx.x * blockDim.x + threadIdx.x;
    int r = i >> 5, c = i & 31;
    if (r < Mn) xp[(size_t)r * 32 + c] = (c < 16) ? x[(size_t)r * 16 + c] : 0.0f;
}

// deg[src[e]] += edge_attr[e]
__global__ void deg_kernel(const int* __restrict__ ei, const float* __restrict__ ea,
                           float* __restrict__ deg, int E) {
    int e = blockIdx.x * blockDim.x + threadIdx.x;
    if (e < E) {
        int s = ei[e];
        __hip_atomic_fetch_add(&deg[s], ea[e], __ATOMIC_RELAXED, __HIP_MEMORY_SCOPE_AGENT);
    }
}

// norm[e] = -rsqrt(deg[src]) * ea * rsqrt(deg[dst])
__global__ void norm_kernel(const int* __restrict__ ei, const float* __restrict__ ea,
                            const float* __restrict__ deg, float* __restrict__ nrm, int E) {
    int e = blockIdx.x * blockDim.x + threadIdx.x;
    if (e < E) {
        int s = ei[e], d = ei[E + e];
        float ds = deg[s], dd = deg[d];
        float rs = ds > 0.f ? rsqrtf(ds) : 0.f;
        float rd = dd > 0.f ? rsqrtf(dd) : 0.f;
        nrm[e] = -rs * ea[e] * rd;
    }
}

// P[dst] += norm[e] * H[src] — one wave per edge; lane handles 4 consecutive
// floats: one global_load_b128 gather + 4 native f32 atomics per lane.
__global__ void prop_kernel(const int* __restrict__ ei, const float* __restrict__ nrm,
                            const float* __restrict__ H, float* __restrict__ P, int E) {
    int lane = threadIdx.x & 31;
    int gw   = (blockIdx.x * blockDim.x + threadIdx.x) >> 5;
    int nw   = (gridDim.x * blockDim.x) >> 5;
    for (int e = gw; e < E; e += nw) {
        int s = ei[e], d = ei[E + e];
        float w = nrm[e];
        size_t bs = (size_t)s * 128 + lane * 4;
        size_t bd = (size_t)d * 128 + lane * 4;
        v4f v = *(const v4f*)&H[bs];
#pragma unroll
        for (int j = 0; j < 4; ++j)
            __hip_atomic_fetch_add(&P[bd + j], w * v[j],
                                   __ATOMIC_RELAXED, __HIP_MEMORY_SCOPE_AGENT);
    }
}

// ---------------------------------------------------------------------------
// Weight pack: f32 row-major (Kact x Nact) -> f16 B-fragment layout
// dst[((kb*NBl+nb)*32 + L)*16 + e] = W[kb*32 + (L/16)*16 + e][nb*16 + L%16]
// ---------------------------------------------------------------------------
__global__ void pack_kernel(const float* __restrict__ W, _Float16* __restrict__ dst,
                            int Kact, int Nact, int NBl, int total) {
    int idx = blockIdx.x * blockDim.x + threadIdx.x;
    if (idx >= total) return;
    int e    = idx & 15;
    int L    = (idx >> 4) & 31;
    int rest = idx >> 9;               // kb*NBl + nb
    int nb   = rest % NBl;
    int kb   = rest / NBl;
    int k = kb * 32 + ((L >> 4) << 4) + e;
    int n = nb * 16 + (L & 15);
    float v = (k < Kact && n < Nact) ? W[(size_t)k * Nact + n] : 0.0f;
    dst[idx] = (_Float16)v;
}

// ---------------------------------------------------------------------------
// Fused GEMM: Out[M x NBL*16] (+)= (ascale*A[M x Kact]) @ Wpk  (+bias,relu,BN-stats)
// 256 threads = 8 waves; block tile 128 rows x NBL*16 cols; wave tile 16 rows.
// Kact must be a multiple of 32 (inputs pre-padded).
// ---------------------------------------------------------------------------
template <int NBL>
__global__ void gemm_wmma(const float* __restrict__ A, int lda, int Kact,
                          const _Float16* __restrict__ Wpk,
                          const float* __restrict__ bias,
                          float* __restrict__ Out, float* __restrict__ stats,
                          int M, int Nact, int accF, int finF, float ascale) {
    constexpr int LDSS = 40;                 // halves; 80B row stride (16B aligned)
    __shared__ __align__(32) _Float16 smem[128 * LDSS];

    const int tid  = threadIdx.x;
    const int lane = tid & 31;
    const int wv   = tid >> 5;
    const int mBase = blockIdx.x * 128;
    const int KB = Kact >> 5;
    const bool fullRows = (mBase + 128 <= M);

    v8f acc[NBL];
#pragma unroll
    for (int i = 0; i < NBL; ++i)
#pragma unroll
        for (int j = 0; j < 8; ++j) acc[i][j] = 0.0f;

    for (int kb = 0; kb < KB; ++kb) {
        if (fullRows) {
            // branchless: 4 x float4 coalesced loads -> packed v4h -> ds_store_b64
#pragma unroll
            for (int i = 0; i < 4; ++i) {
                int row  = (tid >> 3) + (i << 5);
                int col4 = (tid & 7) << 2;
                v4f v = *(const v4f*)&A[(size_t)(mBase + row) * lda + (kb << 5) + col4];
                v4h h;
                h[0] = (_Float16)(v[0] * ascale);
                h[1] = (_Float16)(v[1] * ascale);
                h[2] = (_Float16)(v[2] * ascale);
                h[3] = (_Float16)(v[3] * ascale);
                *(v4h*)&smem[row * LDSS + col4] = h;
            }
        } else {
#pragma unroll
            for (int i = 0; i < 16; ++i) {
                int row = (tid >> 5) + (i << 3);
                int col = tid & 31;
                int gr = mBase + row;
                float v = 0.0f;
                if (gr < M) v = A[(size_t)gr * lda + (kb << 5) + col] * ascale;
                smem[row * LDSS + col] = (_Float16)v;
            }
        }
        __syncthreads();

        // A fragment: lane L, elem e -> M=L%16, K=(L/16)*16 + e
        int arow = (wv << 4) + (lane & 15);
        int hs   = lane >> 4;
        v8h lo = *(const v8h*)&smem[arow * LDSS + hs * 16];
        v8h hi = *(const v8h*)&smem[arow * LDSS + hs * 16 + 8];
        v16h afrag = __builtin_shufflevector(lo, hi, 0, 1, 2, 3, 4, 5, 6, 7,
                                             8, 9, 10, 11, 12, 13, 14, 15);
#pragma unroll
        for (int nb = 0; nb < NBL; ++nb) {
            v16h bfrag = *(const v16h*)&Wpk[(((size_t)kb * NBL + nb) * 32 + lane) * 16];
            acc[nb] = __builtin_amdgcn_wmma_f32_16x16x32_f16(
                false, afrag, false, bfrag, (short)0, acc[nb], false, false);
        }
        __syncthreads();
    }

    // epilogue: C/D layout lane L, vgpr r -> M = r + (L/16)*8, N = L%16
    const int ldo = NBL * 16;
#pragma unroll
    for (int nb = 0; nb < NBL; ++nb) {
        int col = nb * 16 + (lane & 15);
        if (col >= Nact) continue;
        float s = 0.0f, s2 = 0.0f;
#pragma unroll
        for (int r = 0; r < 8; ++r) {
            int grow = mBase + (wv << 4) + r + ((lane >> 4) << 3);
            if (grow < M) {
                float v = acc[nb][r];
                size_t oi = (size_t)grow * ldo + col;
                if (accF) v += Out[oi];
                if (finF) {
                    v += bias[col];
                    v = fmaxf(v, 0.0f);
                    s += v; s2 += v * v;
                }
                Out[oi] = v;
            }
        }
        if (finF) {
            __hip_atomic_fetch_add(&stats[col], s, __ATOMIC_RELAXED, __HIP_MEMORY_SCOPE_AGENT);
            __hip_atomic_fetch_add(&stats[ldo + col], s2, __ATOMIC_RELAXED, __HIP_MEMORY_SCOPE_AGENT);
        }
    }
}

// scale = g*rsqrt(var+eps); shift = beta - mean*scale
__global__ void bn_scale(const float* __restrict__ stats, int stride,
                         const float* __restrict__ g, const float* __restrict__ beta,
                         float* __restrict__ scale, float* __restrict__ shift,
                         int Cact, float invM) {
    int c = blockIdx.x * blockDim.x + threadIdx.x;
    if (c < Cact) {
        float mean = stats[c] * invM;
        float var  = stats[stride + c] * invM - mean * mean;
        float sc = g[c] * rsqrtf(var + EPS_BN);
        scale[c] = sc;
        shift[c] = beta[c] - mean * sc;
    }
}

__global__ void bn_apply(const float* __restrict__ U, int ldu,
                         const float* __restrict__ scale, const float* __restrict__ shift,
                         float* __restrict__ H, int ldh, int M, int Cact) {
    size_t i = (size_t)blockIdx.x * blockDim.x + threadIdx.x;
    size_t total = (size_t)M * Cact;
    if (i < total) {
        int r = (int)(i / Cact);
        int c = (int)(i - (size_t)r * Cact);
        H[(size_t)r * ldh + c] = U[(size_t)r * ldu + c] * scale[c] + shift[c];
    }
}

// ---------------------------------------------------------------------------
// Host driver
// ---------------------------------------------------------------------------
extern "C" void kernel_launch(void* const* d_in, const int* in_sizes, int n_in,
                              void* d_out, int out_size, void* d_ws, size_t ws_size,
                              hipStream_t stream) {
    (void)n_in; (void)out_size; (void)ws_size;
    const int HLD = 128, IDIM = 16, ODIM = 8;

    // ---- resolve input layout (three plausible flatten conventions) ----
    const int E = in_sizes[1] / 2;                 // edge_index is always slot 1
    int xi, eai, pbase; bool sortedParams;
    if (in_sizes[2] == E) {                        // top-level insertion order
        xi = 0; eai = 2; pbase = 3;
        sortedParams = (in_sizes[3] != IDIM * HLD);  // 2048 => insertion-order params
    } else {                                       // fully sorted tree flatten
        eai = 0; pbase = 2; sortedParams = true;
        xi = 2 + 40;                               // 40 param leaves, x last
    }
    const int Mn = in_sizes[xi] / IDIM;

    const float* x   = (const float*)d_in[xi];
    const int*   ei  = (const int*)d_in[1];
    const float* ea  = (const float*)d_in[eai];

    const float *encW[3], *encB[3], *encG[3], *encBt[3];
    const float *cvL[4], *cvB[4], *cvG[4], *cvBt[4];
    const float *decW[3], *decB[3], *decG[3], *decBt[3];
    if (sortedParams) {
        for (int c = 0; c < 4; ++c) { int b = pbase + 4 * c;
            cvBt[c] = (const float*)d_in[b + 0]; cvB[c] = (const float*)d_in[b + 1];
            cvG[c]  = (const float*)d_in[b + 2]; cvL[c] = (const float*)d_in[b + 3]; }
        for (int l = 0; l < 3; ++l) { int b = pbase + 16 + 4 * l;
            decB[l] = (const float*)d_in[b + 0]; decBt[l] = (const float*)d_in[b + 1];
            decG[l] = (const float*)d_in[b + 2]; decW[l]  = (const float*)d_in[b + 3]; }
        for (int l = 0; l < 3; ++l) { int b = pbase + 28 + 4 * l;
            encB[l] = (const float*)d_in[b + 0]; encBt[l] = (const float*)d_in[b + 1];
            encG[l] = (const float*)d_in[b + 2]; encW[l]  = (const float*)d_in[b + 3]; }
    } else {
        for (int l = 0; l < 3; ++l) { int b = pbase + 4 * l;
            encW[l] = (const float*)d_in[b + 0]; encB[l] = (const float*)d_in[b + 1];
            encG[l] = (const float*)d_in[b + 2]; encBt[l] = (const float*)d_in[b + 3]; }
        for (int c = 0; c < 4; ++c) { int b = pbase + 12 + 4 * c;
            cvL[c] = (const float*)d_in[b + 0]; cvB[c] = (const float*)d_in[b + 1];
            cvG[c] = (const float*)d_in[b + 2]; cvBt[c] = (const float*)d_in[b + 3]; }
        for (int l = 0; l < 3; ++l) { int b = pbase + 24 + 4 * l;
            decW[l] = (const float*)d_in[b + 0]; decB[l] = (const float*)d_in[b + 1];
            decG[l] = (const float*)d_in[b + 2]; decBt[l] = (const float*)d_in[b + 3]; }
    }

    // ---- workspace carve-up (everything stays L2-resident: ~92 MB) ----
    const size_t MC = (size_t)Mn * HLD;
    float* H   = (float*)d_ws;
    float* U   = H + MC;
    float* T1  = U + MC;
    float* T2  = T1 + MC;
    float* Xp  = T2 + MC;                // Mn x 32 padded input
    float* deg = Xp + (size_t)Mn * 32;
    float* nrm = deg + Mn;
    float* stats = nrm + E;              // 256 floats
    float* scl = stats + 256;            // 128
    float* shf = scl + 128;              // 128
    _Float16* pk = (_Float16*)(shf + 128);
    _Float16* pkE[3]; _Float16* pkC[4][3]; _Float16* pkD[3];
    pkE[0] = pk; pk += 1 * 8 * 512;
    pkE[1] = pk; pk += 4 * 8 * 512;
    pkE[2] = pk; pk += 4 * 8 * 512;
    for (int c = 0; c < 4; ++c)
        for (int j = 0; j < 3; ++j) { pkC[c][j] = pk; pk += 4 * 8 * 512; }
    pkD[0] = pk; pk += 4 * 8 * 512;
    pkD[1] = pk; pk += 4 * 8 * 512;
    pkD[2] = pk; pk += 4 * 1 * 512;

    auto cdiv = [](size_t a, size_t b) { return (int)((a + b - 1) / b); };
    auto pack = [&](const float* W, _Float16* dst, int Kact, int Nact, int KB, int NBl) {
        int total = KB * NBl * 512;
        pack_kernel<<<cdiv(total, 256), 256, 0, stream>>>(W, dst, Kact, Nact, NBl, total);
    };

    // ---- pack all weights; pad x ----
    pack(encW[0], pkE[0], IDIM, HLD, 1, 8);
    pack(encW[1], pkE[1], HLD, HLD, 4, 8);
    pack(encW[2], pkE[2], HLD, HLD, 4, 8);
    for (int c = 0; c < 4; ++c)
        for (int j = 0; j < 3; ++j)
            pack(cvL[c] + (size_t)j * HLD * HLD, pkC[c][j], HLD, HLD, 4, 8);
    pack(decW[0], pkD[0], HLD, HLD, 4, 8);
    pack(decW[1], pkD[1], HLD, HLD, 4, 8);
    pack(decW[2], pkD[2], HLD, ODIM, 4, 1);
    pad_x<<<cdiv((size_t)Mn * 32, 256), 256, 0, stream>>>(x, Xp, Mn);

    // ---- graph normalization ----
    zero_f32<<<cdiv(Mn, 256), 256, 0, stream>>>(deg, Mn);
    deg_kernel<<<cdiv(E, 256), 256, 0, stream>>>(ei, ea, deg, E);
    norm_kernel<<<cdiv(E, 256), 256, 0, stream>>>(ei, ea, deg, nrm, E);

    const int gGemm = cdiv(Mn, 128);
    const float invM = 1.0f / (float)Mn;
    const int gProp = 2048;

    auto dense_bn = [&](const float* A, int lda, int Kact, _Float16* W,
                        const float* b, const float* g, const float* bt, float* out) {
        zero_f32<<<1, 256, 0, stream>>>(stats, 256);
        gemm_wmma<8><<<gGemm, 256, 0, stream>>>(A, lda, Kact, W, b, U, stats,
                                                Mn, HLD, 0, 1, 1.0f);
        bn_scale<<<1, 128, 0, stream>>>(stats, 128, g, bt, scl, shf, HLD, invM);
        bn_apply<<<cdiv(MC, 256), 256, 0, stream>>>(U, HLD, scl, shf, out, HLD, Mn, HLD);
    };

    // ---- encoder ----
    dense_bn(Xp, 32, 32, pkE[0], encB[0], encG[0], encBt[0], H);
    dense_bn(H, HLD, HLD, pkE[1], encB[1], encG[1], encBt[1], H);
    dense_bn(H, HLD, HLD, pkE[2], encB[2], encG[2], encBt[2], H);

    // ---- ChebConv layers (K=3) ----
    for (int c = 0; c < 4; ++c) {
        gemm_wmma<8><<<gGemm, 256, 0, stream>>>(H, HLD, HLD, pkC[c][0], nullptr, U,
                                                nullptr, Mn, HLD, 0, 0, 1.0f);
        zero_f32<<<cdiv(MC, 256), 256, 0, stream>>>(T1, MC);
        prop_kernel<<<gProp, 256, 0, stream>>>(ei, nrm, H, T1, E);
        gemm_wmma<8><<<gGemm, 256, 0, stream>>>(T1, HLD, HLD, pkC[c][1], nullptr, U,
                                                nullptr, Mn, HLD, 1, 0, 1.0f);
        scale_init<<<cdiv(MC, 256), 256, 0, stream>>>(H, T2, MC);
        prop_kernel<<<gProp, 256, 0, stream>>>(ei, nrm, T1, T2, E);
        zero_f32<<<1, 256, 0, stream>>>(stats, 256);
        gemm_wmma<8><<<gGemm, 256, 0, stream>>>(T2, HLD, HLD, pkC[c][2], cvB[c], U,
                                                stats, Mn, HLD, 1, 1, 2.0f);
        bn_scale<<<1, 128, 0, stream>>>(stats, 128, cvG[c], cvBt[c], scl, shf, HLD, invM);
        bn_apply<<<cdiv(MC, 256), 256, 0, stream>>>(U, HLD, scl, shf, H, HLD, Mn, HLD);
    }

    // ---- decoder ----
    dense_bn(H, HLD, HLD, pkD[0], decB[0], decG[0], decBt[0], H);
    dense_bn(H, HLD, HLD, pkD[1], decB[1], decG[1], decBt[1], H);
    zero_f32<<<1, 256, 0, stream>>>(stats, 256);
    gemm_wmma<1><<<gGemm, 256, 0, stream>>>(H, HLD, HLD, pkD[2], decB[2], T1, stats,
                                            Mn, ODIM, 0, 1, 1.0f);
    bn_scale<<<1, 128, 0, stream>>>(stats, 16, decG[2], decBt[2], scl, shf, ODIM, invM);
    bn_apply<<<cdiv((size_t)Mn * ODIM, 256), 256, 0, stream>>>(T1, 16, scl, shf,
                                                               (float*)d_out, ODIM, Mn, ODIM);
}
